// TransformerBlock_72730976191104
// MI455X (gfx1250) — compile-verified, matrix operand order checked by reading, not run on hardware
//
#include <hip/hip_runtime.h>

// ---------------------------------------------------------------------------
// Problem constants (match the JAX reference)
// ---------------------------------------------------------------------------
#define BB 4
#define TT 4096
#define EE 1024
#define HH 16
#define HD 64
#define MM 64
#define ROWS (BB * TT)      // 16384
#define STAGES 10           // log2(E)
#define PI_F 3.14159265358979323846f

typedef __bf16 v16bf __attribute__((ext_vector_type(16)));
typedef float  v8f   __attribute__((ext_vector_type(8)));
typedef unsigned int u32x4 __attribute__((ext_vector_type(4)));
typedef int i32x4 __attribute__((ext_vector_type(4)));
typedef int i32x8 __attribute__((ext_vector_type(8)));

#if defined(__gfx1250__) && __has_builtin(__builtin_amdgcn_tensor_load_to_lds)
#define HAVE_TDM 1
#else
#define HAVE_TDM 0
#endif

__device__ __forceinline__ unsigned short f2bf(float f) {
    union { float f; unsigned u; } v; v.f = f;
    unsigned r = (v.u + 0x7FFFu + ((v.u >> 16) & 1u)) >> 16;
    return (unsigned short)r;
}

#if HAVE_TDM
// ---------------------------------------------------------------------------
// Issue a 2D TDM tile load: global (row-major, rows of `stride0` dwords)
// -> LDS at byte offset lds_off. Tile = tile_d0 dwords x tile_d1 rows.
// D# per CDNA5 ISA 8.3/8.4 (data_size=4B, 2D: VADDR2/3 groups zeroed).
// ---------------------------------------------------------------------------
__device__ __forceinline__ void tdm_load_2d(unsigned lds_off, const void* gptr,
                                            unsigned tensor_d0, unsigned tensor_d1,
                                            unsigned tile_d0, unsigned tile_d1,
                                            unsigned stride0) {
    unsigned long long ga = (unsigned long long)(uintptr_t)gptr;
    u32x4 g0;
    g0.x = 1u;                                     // count=1, user mode
    g0.y = lds_off;                                // lds_addr
    g0.z = (unsigned)(ga & 0xffffffffu);           // global_addr[31:0]
    g0.w = (unsigned)((ga >> 32) & 0x1ffffffu)     // global_addr[56:32]
         | (2u << 30);                             // type=2 ("image")
    i32x8 g1;
    g1[0] = (int)(2u << 16);                                        // data_size=4B
    g1[1] = (int)((tensor_d0 & 0xffffu) << 16);                     // dim0[15:0]
    g1[2] = (int)((tensor_d0 >> 16) | ((tensor_d1 & 0xffffu) << 16));
    g1[3] = (int)((tensor_d1 >> 16) | (tile_d0 << 16));             // tile_dim0
    g1[4] = (int)(tile_d1 & 0xffffu);                               // tile_dim1
    g1[5] = (int)stride0;                                           // dim0_stride
    g1[6] = 0;
    g1[7] = 0;
    i32x4 z4 = {0, 0, 0, 0};
#if __clang_major__ >= 23
    i32x8 z8 = {0, 0, 0, 0, 0, 0, 0, 0};
    __builtin_amdgcn_tensor_load_to_lds(g0, g1, z4, z4, z8, 0);
#else
    __builtin_amdgcn_tensor_load_to_lds(g0, g1, z4, z4, 0);
#endif
}
#endif

// ---------------------------------------------------------------------------
// RMSNorm (x, w) -> bf16 activations, one block per row, 256 threads
// ---------------------------------------------------------------------------
__global__ void rmsnorm_bf16_kernel(const float* __restrict__ x,
                                    const float* __restrict__ w,
                                    unsigned short* __restrict__ xn) {
    int t = blockIdx.x, b = blockIdx.y, tid = threadIdx.x;
    long row = (long)(b * TT + t) * EE;
    __shared__ float red[256];
    __shared__ float srms;
    float v[4]; float ss = 0.f;
#pragma unroll
    for (int r = 0; r < 4; ++r) { v[r] = x[row + tid + r * 256]; ss += v[r] * v[r]; }
    red[tid] = ss; __syncthreads();
    for (int off = 128; off; off >>= 1) {
        if (tid < off) red[tid] += red[tid + off];
        __syncthreads();
    }
    if (tid == 0) srms = rsqrtf(red[0] / (float)EE + 1e-6f);
    __syncthreads();
#pragma unroll
    for (int r = 0; r < 4; ++r) {
        int e = tid + r * 256;
        xn[row + e] = f2bf(v[r] * srms * w[e]);
    }
}

// ---------------------------------------------------------------------------
// Weight transpose + fp32->bf16: W[K][N] -> Wt[N][K], 32x32 LDS tiles
// grid (E/32, E/32), 256 threads
// ---------------------------------------------------------------------------
__global__ void transpose_bf16_kernel(const float* __restrict__ W,
                                      unsigned short* __restrict__ Wt) {
    __shared__ float tile[32][33];
    int bn = blockIdx.x * 32;   // n
    int bk = blockIdx.y * 32;   // k
    int tx = threadIdx.x & 31, ty = threadIdx.x >> 5;   // ty: 0..7
#pragma unroll
    for (int r = 0; r < 32; r += 8)
        tile[ty + r][tx] = W[(long)(bk + ty + r) * EE + bn + tx];
    __syncthreads();
#pragma unroll
    for (int r = 0; r < 32; r += 8)
        Wt[(long)(bn + ty + r) * EE + bk + tx] = f2bf(tile[tx][ty + r]);
}

// ---------------------------------------------------------------------------
// bf16 WMMA GEMM: C[M,N] = A[M,K] * Wt[N,K]^T + bias  (both operands K-major)
// 128x128 block tile, TK=32, 256 threads (8 waves), each wave 32x64
// (2x4 grid of 16x16x32 bf16 WMMA tiles). Tiles staged into LDS by the
// Tensor Data Mover (one issue per workgroup from wave 0), double buffered,
// synchronized with s_wait_tensorcnt + workgroup barrier.
// ---------------------------------------------------------------------------
#define GTM 128
#define GTN 128
#define GTK 32

__global__ __launch_bounds__(256)
void gemm_bf16_wmma_kernel(const unsigned short* __restrict__ A,
                           const unsigned short* __restrict__ Wt,
                           const float* __restrict__ bias,
                           float* __restrict__ C,
                           int Mrows, int N, int K) {
    __shared__ unsigned short As[2][GTM * GTK];
    __shared__ unsigned short Bs[2][GTN * GTK];   // [n][k]

    int tid  = threadIdx.x;
    int lane = tid & 31;
    int w    = tid >> 5;       // wave 0..7
    int wr   = w & 3;          // row group (4)  -> 32 rows each
    int wc   = w >> 2;         // col group (2)  -> 64 cols each
    int lr   = lane & 15;
    int lh   = (lane >> 4) & 1;
    int m0   = blockIdx.y * GTM;
    int n0   = blockIdx.x * GTN;

    v8f acc[2][4];
#pragma unroll
    for (int i = 0; i < 2; ++i)
#pragma unroll
        for (int j = 0; j < 4; ++j)
            acc[i][j] = (v8f){0.f, 0.f, 0.f, 0.f, 0.f, 0.f, 0.f, 0.f};

#if HAVE_TDM
    auto issue_tiles = [&](int buf, int kt) {
        if (w == 0) {   // wave-uniform: exactly one wave issues the DMAs
            tdm_load_2d((unsigned)(uintptr_t)&As[buf][0],
                        A + (long)m0 * K + kt,
                        (unsigned)(K / 2), (unsigned)Mrows,
                        GTK / 2, GTM, (unsigned)(K / 2));
            tdm_load_2d((unsigned)(uintptr_t)&Bs[buf][0],
                        Wt + (long)n0 * K + kt,
                        (unsigned)(K / 2), (unsigned)N,
                        GTK / 2, GTN, (unsigned)(K / 2));
        }
    };
#else
    auto issue_tiles = [&](int buf, int kt) {
#pragma unroll
        for (int c2 = 0; c2 < 2; ++c2) {
            int ch = tid * 2 + c2;          // 512 chunks of 8 bf16
            int r = ch >> 2, c = ch & 3;
            *(uint4*)&As[buf][r * GTK + c * 8] =
                *(const uint4*)&A[(long)(m0 + r) * K + kt + c * 8];
            *(uint4*)&Bs[buf][r * GTK + c * 8] =
                *(const uint4*)&Wt[(long)(n0 + r) * K + kt + c * 8];
        }
    };
#endif

    auto wait_tiles = [&]() {
#if HAVE_TDM
        __builtin_amdgcn_s_wait_tensorcnt(0);   // no-op for non-issuing waves
#endif
        __syncthreads();
    };

    issue_tiles(0, 0);
    wait_tiles();

    for (int kt = 0; kt < K; kt += GTK) {
        int buf = (kt / GTK) & 1;
        if (kt + GTK < K) issue_tiles(buf ^ 1, kt + GTK);

        v16bf afrag[2], bfrag[4];
#pragma unroll
        for (int mi = 0; mi < 2; ++mi)
            afrag[mi] = *(const v16bf*)&As[buf][(wr * 32 + mi * 16 + lr) * GTK + lh * 16];
#pragma unroll
        for (int ni = 0; ni < 4; ++ni)
            bfrag[ni] = *(const v16bf*)&Bs[buf][(wc * 64 + ni * 16 + lr) * GTK + lh * 16];
#pragma unroll
        for (int mi = 0; mi < 2; ++mi)
#pragma unroll
            for (int ni = 0; ni < 4; ++ni)
                acc[mi][ni] = __builtin_amdgcn_wmma_f32_16x16x32_bf16(
                    false, afrag[mi], false, bfrag[ni],
                    (short)0, acc[mi][ni], false, false);
        wait_tiles();
    }

#pragma unroll
    for (int mi = 0; mi < 2; ++mi)
#pragma unroll
        for (int ni = 0; ni < 4; ++ni)
#pragma unroll
            for (int r = 0; r < 8; ++r) {
                int row = m0 + wr * 32 + mi * 16 + lh * 8 + r;
                int col = n0 + wc * 64 + ni * 16 + lr;
                C[(long)row * N + col] = acc[mi][ni][r] + bias[col];
            }
}

// ---------------------------------------------------------------------------
// RoPE + transpose [B,T,E] -> [B,H,T,HD]; do_rope=0 is a plain transpose (v)
// grid (T, B), 512 threads: thread = (head h, pair i2)
// ---------------------------------------------------------------------------
__global__ void rope_transpose_kernel(const float* __restrict__ src,
                                      float* __restrict__ dst, int do_rope) {
    int t = blockIdx.x, b = blockIdx.y, tid = threadIdx.x;
    int h  = tid >> 5;
    int i2 = tid & 31;
    long sidx = (long)(b * TT + t) * EE + h * HD + 2 * i2;
    float e0 = src[sidx], e1 = src[sidx + 1];
    long didx = ((long)(b * HH + h) * TT + t) * HD + 2 * i2;
    if (do_rope) {
        float inv = __powf(10000.f, -(float)(2 * i2) / (float)HD);
        float ang = (float)t * inv;
        float c = __cosf(ang), s = __sinf(ang);
        dst[didx]     = e0 * c - e1 * s;
        dst[didx + 1] = e1 * c + e0 * s;
    } else {
        dst[didx]     = e0;
        dst[didx + 1] = e1;
    }
}

// ---------------------------------------------------------------------------
// Landmarks: mean over contiguous chunks of T/M rows. grid (M,H,B), 64 threads
// ---------------------------------------------------------------------------
__global__ void landmark_kernel(const float* __restrict__ src,
                                float* __restrict__ lmk) {
    int j = blockIdx.x, h = blockIdx.y, b = blockIdx.z, d = threadIdx.x;
    int bh = b * HH + h;
    const int seg = TT / MM;  // 64
    float s = 0.f;
    for (int u = 0; u < seg; ++u)
        s += src[((long)bh * TT + j * seg + u) * HD + d];
    lmk[((long)bh * MM + j) * HD + d] = s * (1.f / (float)seg);
}

// ---------------------------------------------------------------------------
// G = softmax_t(q_l . k^T) @ v  (flash-style over T) — one block per (b,h,i)
// ---------------------------------------------------------------------------
__global__ __launch_bounds__(256)
void f3v_kernel(const float* __restrict__ ql,
                const float* __restrict__ kT,
                const float* __restrict__ vT,
                float* __restrict__ G) {
    int i = blockIdx.x, h = blockIdx.y, b = blockIdx.z, tid = threadIdx.x;
    int bh = b * HH + h;
    __shared__ float qrow[HD], accv[HD], ps[256], red[256];
    __shared__ float sm, sl;
    if (tid < HD) { qrow[tid] = ql[((long)bh * MM + i) * HD + tid]; accv[tid] = 0.f; }
    if (tid == 0) { sm = -1e30f; sl = 0.f; }
    __syncthreads();
    const float scale = 0.125f;  // 1/sqrt(64)
    for (int tile = 0; tile < TT / 256; ++tile) {
        int t = tile * 256 + tid;
        const float* kp = &kT[((long)bh * TT + t) * HD];
        float s = 0.f;
        for (int d = 0; d < HD; ++d) s += qrow[d] * kp[d];
        s *= scale;
        red[tid] = s; __syncthreads();
        for (int off = 128; off; off >>= 1) {
            if (tid < off) red[tid] = fmaxf(red[tid], red[tid + off]);
            __syncthreads();
        }
        float mnew  = fmaxf(sm, red[0]);
        float alpha = __expf(sm - mnew);
        __syncthreads();
        float p = __expf(s - mnew);
        ps[tid] = p; red[tid] = p; __syncthreads();
        for (int off = 128; off; off >>= 1) {
            if (tid < off) red[tid] += red[tid + off];
            __syncthreads();
        }
        if (tid < HD) {
            float a = accv[tid] * alpha;
            const float* vp = &vT[((long)bh * TT + tile * 256) * HD + tid];
            for (int j = 0; j < 256; ++j) a += ps[j] * vp[(long)j * HD];
            accv[tid] = a;
        }
        if (tid == 0) { sl = sl * alpha + red[0]; sm = mnew; }
        __syncthreads();
    }
    if (tid < HD) G[((long)bh * MM + i) * HD + tid] = accv[tid] / sl;
}

// ---------------------------------------------------------------------------
// 64x64 matmul helper (generic pointers, 256 threads, 16 outs/thread)
// ---------------------------------------------------------------------------
__device__ void mm64(const float* A, const float* B, float* C, int tid) {
    for (int idx = tid; idx < MM * MM; idx += 256) {
        int i = idx >> 6, j = idx & 63;
        float s = 0.f;
        for (int k = 0; k < MM; ++k) s += A[i * MM + k] * B[k * MM + j];
        C[idx] = s;
    }
}

// ---------------------------------------------------------------------------
// Per (b,h): F2 = softmax(q_l k_l^T * scale); V = iterative_pinv(F2);
// P = V @ G.   grid (H,B), 256 threads, global scratch t1/t2/t3 per block.
// ---------------------------------------------------------------------------
__global__ __launch_bounds__(256)
void pinv_kernel(const float* __restrict__ ql,
                 const float* __restrict__ kl,
                 const float* __restrict__ G,
                 float* __restrict__ P,
                 float* __restrict__ scratch) {
    int h = blockIdx.x, b = blockIdx.y, tid = threadIdx.x;
    int bh = b * HH + h;
    __shared__ float Km[MM * MM];
    __shared__ float Vm[MM * MM];
    __shared__ float rs[MM], cs[MM];
    __shared__ float snorm;
    float* t1 = scratch + (long)bh * 3 * MM * MM;
    float* t2 = t1 + MM * MM;
    float* t3 = t2 + MM * MM;
    const float scale = 0.125f;

    for (int idx = tid; idx < MM * MM; idx += 256) {
        int i = idx >> 6, j = idx & 63;
        float s = 0.f;
        for (int d = 0; d < HD; ++d)
            s += ql[((long)bh * MM + i) * HD + d] * kl[((long)bh * MM + j) * HD + d];
        Km[idx] = s * scale;
    }
    __syncthreads();
    if (tid < MM) {
        float mx = -1e30f;
        for (int j = 0; j < MM; ++j) mx = fmaxf(mx, Km[tid * MM + j]);
        float s = 0.f;
        for (int j = 0; j < MM; ++j) { float e = __expf(Km[tid * MM + j] - mx); Km[tid * MM + j] = e; s += e; }
        float inv = 1.f / s;
        for (int j = 0; j < MM; ++j) Km[tid * MM + j] *= inv;
    }
    __syncthreads();
    if (tid < MM) {
        float r = 0.f, c = 0.f;
        for (int j = 0; j < MM; ++j) { r += fabsf(Km[tid * MM + j]); c += fabsf(Km[j * MM + tid]); }
        rs[tid] = r; cs[tid] = c;
    }
    __syncthreads();
    if (tid == 0) {
        float rm = 0.f, cm = 0.f;
        for (int j = 0; j < MM; ++j) { rm = fmaxf(rm, rs[j]); cm = fmaxf(cm, cs[j]); }
        snorm = 1.f / (rm * cm);
    }
    __syncthreads();
    for (int idx = tid; idx < MM * MM; idx += 256) {
        int i = idx >> 6, j = idx & 63;
        Vm[idx] = Km[j * MM + i] * snorm;
    }
    __syncthreads();

    for (int it = 0; it < 6; ++it) {
        mm64(Km, Vm, t1, tid); __syncthreads();                       // KV
        for (int idx = tid; idx < MM * MM; idx += 256) {
            int i = idx >> 6, j = idx & 63;
            t2[idx] = ((i == j) ? 7.f : 0.f) - t1[idx];
        }
        __syncthreads();
        mm64(t1, t2, t3, tid); __syncthreads();                       // KV(7I-KV)
        for (int idx = tid; idx < MM * MM; idx += 256) {
            int i = idx >> 6, j = idx & 63;
            t2[idx] = ((i == j) ? 15.f : 0.f) - t3[idx];
        }
        __syncthreads();
        mm64(t1, t2, t3, tid); __syncthreads();                       // KV(15I-..)
        for (int idx = tid; idx < MM * MM; idx += 256) {
            int i = idx >> 6, j = idx & 63;
            t2[idx] = ((i == j) ? 13.f : 0.f) - t3[idx];
        }
        __syncthreads();
        mm64(Vm, t2, t3, tid); __syncthreads();                       // V(13I-..)
        for (int idx = tid; idx < MM * MM; idx += 256) Vm[idx] = 0.25f * t3[idx];
        __syncthreads();
    }
    for (int idx = tid; idx < MM * MM; idx += 256) t1[idx] = G[(long)bh * MM * MM + idx];
    __syncthreads();
    mm64(Vm, t1, t2, tid); __syncthreads();
    for (int idx = tid; idx < MM * MM; idx += 256) P[(long)bh * MM * MM + idx] = t2[idx];
}

// ---------------------------------------------------------------------------
// attn = softmax_m(q k_l^T) @ P, written as bf16 in [B,T,E] layout
// grid (T/256, H, B), 256 threads (one t per thread)
// ---------------------------------------------------------------------------
__global__ __launch_bounds__(256)
void f1p_kernel(const float* __restrict__ qT,
                const float* __restrict__ kl,
                const float* __restrict__ P,
                unsigned short* __restrict__ attnC) {
    int tb = blockIdx.x, h = blockIdx.y, b = blockIdx.z, tid = threadIdx.x;
    int bh = b * HH + h;
    __shared__ float kls[MM * HD];
    __shared__ float Ps[MM * HD];
    for (int idx = tid; idx < MM * HD; idx += 256) {
        kls[idx] = kl[(long)bh * MM * HD + idx];
        Ps[idx]  = P[(long)bh * MM * HD + idx];
    }
    __syncthreads();
    int t = tb * 256 + tid;
    const float* qp = &qT[((long)bh * TT + t) * HD];
    float qv[HD];
    for (int d = 0; d < HD; ++d) qv[d] = qp[d];
    const float scale = 0.125f;
    float p[MM]; float mx = -1e30f;
    for (int j = 0; j < MM; ++j) {
        float s = 0.f;
        for (int d = 0; d < HD; ++d) s += qv[d] * kls[j * HD + d];
        s *= scale;
        p[j] = s; mx = fmaxf(mx, s);
    }
    float sum = 0.f;
    for (int j = 0; j < MM; ++j) { p[j] = __expf(p[j] - mx); sum += p[j]; }
    float inv = 1.f / sum;
    for (int d = 0; d < HD; ++d) {
        float o = 0.f;
        for (int j = 0; j < MM; ++j) o += p[j] * Ps[j * HD + d];
        attnC[((long)(b * TT + t)) * EE + h * HD + d] = f2bf(o * inv);
    }
}

// ---------------------------------------------------------------------------
// Gate: z = butterfly(x, gate_thetas) + gate_bias; g from arctan; mix h and x
// grid (T,B), 512 threads
// ---------------------------------------------------------------------------
__global__ __launch_bounds__(512)
void gate_kernel(const float* __restrict__ x,
                 const float* __restrict__ hbuf,
                 const float* __restrict__ alpha,
                 const float* __restrict__ gthetas,
                 const float* __restrict__ gbias,
                 float* __restrict__ x2) {
    int t = blockIdx.x, b = blockIdx.y, tid = threadIdx.x;
    long row = (long)(b * TT + t) * EE;
    __shared__ float z[EE];
    z[tid] = x[row + tid];
    z[tid + 512] = x[row + tid + 512];
    __syncthreads();
    for (int s = 0; s < STAGES; ++s) {
        int bw = 1 << s;
        int j  = tid & (bw - 1);
        int i0 = 2 * (tid - j) + j;
        int i1 = i0 + bw;
        float th = gthetas[s * (EE / 2) + tid];
        float c = __cosf(th), si = __sinf(th);
        float a0 = z[i0], a1 = z[i1];
        z[i0] = c * a0 - si * a1;
        z[i1] = si * a0 + c * a1;
        __syncthreads();
    }
#pragma unroll
    for (int r = 0; r < 2; ++r) {
        int e = tid + r * 512;
        float zz   = z[e] + gbias[e];
        float base = (atanf(zz) + PI_F * 0.5f) * (1.f / PI_F);
        float al   = alpha[e];
        float g    = base * (1.f + 2.f * al) - al;
        x2[row + e] = hbuf[row + e] * g + x[row + e] * (1.f - g);
    }
}

// ---------------------------------------------------------------------------
// FFN: h2 = butterfly(rmsnorm(x2, ln2), ffn_thetas); out = x2 + h2
// grid (T,B), 512 threads
// ---------------------------------------------------------------------------
__global__ __launch_bounds__(512)
void ffn_kernel(const float* __restrict__ x2,
                const float* __restrict__ ln2,
                const float* __restrict__ fthetas,
                float* __restrict__ out) {
    int t = blockIdx.x, b = blockIdx.y, tid = threadIdx.x;
    long row = (long)(b * TT + t) * EE;
    __shared__ float z[EE];
    __shared__ float red[512];
    __shared__ float srms;
    float a0 = x2[row + tid], a1 = x2[row + tid + 512];
    red[tid] = a0 * a0 + a1 * a1; __syncthreads();
    for (int off = 256; off; off >>= 1) {
        if (tid < off) red[tid] += red[tid + off];
        __syncthreads();
    }
    if (tid == 0) srms = rsqrtf(red[0] / (float)EE + 1e-6f);
    __syncthreads();
    z[tid]       = a0 * srms * ln2[tid];
    z[tid + 512] = a1 * srms * ln2[tid + 512];
    __syncthreads();
    for (int s = 0; s < STAGES; ++s) {
        int bw = 1 << s;
        int j  = tid & (bw - 1);
        int i0 = 2 * (tid - j) + j;
        int i1 = i0 + bw;
        float th = fthetas[s * (EE / 2) + tid];
        float c = __cosf(th), si = __sinf(th);
        float v0 = z[i0], v1 = z[i1];
        z[i0] = c * v0 - si * v1;
        z[i1] = si * v0 + c * v1;
        __syncthreads();
    }
    out[row + tid]       = a0 + z[tid];
    out[row + tid + 512] = a1 + z[tid + 512];
}

// ---------------------------------------------------------------------------
// Host-side launcher
// ---------------------------------------------------------------------------
extern "C" void kernel_launch(void* const* d_in, const int* in_sizes, int n_in,
                              void* d_out, int out_size, void* d_ws, size_t ws_size,
                              hipStream_t stream) {
    const float* x       = (const float*)d_in[0];
    const float* ln1_w   = (const float*)d_in[1];
    const float* wq      = (const float*)d_in[2];
    const float* bq      = (const float*)d_in[3];
    const float* wk      = (const float*)d_in[4];
    const float* bk      = (const float*)d_in[5];
    const float* wv      = (const float*)d_in[6];
    const float* bv      = (const float*)d_in[7];
    const float* wo      = (const float*)d_in[8];
    const float* bo      = (const float*)d_in[9];
    const float* alpha   = (const float*)d_in[10];
    const float* gthetas = (const float*)d_in[11];
    const float* gbias   = (const float*)d_in[12];
    const float* fthetas = (const float*)d_in[13];
    const float* ln2_w   = (const float*)d_in[14];
    float* out = (float*)d_out;

    // ---- workspace carve-up ----
    char* ws = (char*)d_ws;
    size_t off = 0;
    auto carve = [&](size_t bytes) -> void* {
        void* p = ws + off;
        off += (bytes + 255) & ~(size_t)255;
        return p;
    };
    unsigned short* xn   = (unsigned short*)carve((size_t)ROWS * EE * 2);
    unsigned short* wqb  = (unsigned short*)carve((size_t)EE * EE * 2);
    unsigned short* wkb  = (unsigned short*)carve((size_t)EE * EE * 2);
    unsigned short* wvb  = (unsigned short*)carve((size_t)EE * EE * 2);
    unsigned short* wob  = (unsigned short*)carve((size_t)EE * EE * 2);
    float* tmp           = (float*)carve((size_t)ROWS * EE * 4);  // GEMM out / h
    float* qT            = (float*)carve((size_t)BB * HH * TT * HD * 4);
    float* kT            = (float*)carve((size_t)BB * HH * TT * HD * 4);
    float* vT            = (float*)carve((size_t)BB * HH * TT * HD * 4);
    float* ql            = (float*)carve((size_t)BB * HH * MM * HD * 4);
    float* kl            = (float*)carve((size_t)BB * HH * MM * HD * 4);
    float* G             = (float*)carve((size_t)BB * HH * MM * HD * 4);
    float* P             = (float*)carve((size_t)BB * HH * MM * HD * 4);
    float* pscr          = (float*)carve((size_t)BB * HH * 3 * MM * MM * 4);
    unsigned short* attnC = (unsigned short*)carve((size_t)ROWS * EE * 2);
    float* x2            = (float*)carve((size_t)ROWS * EE * 4);

    // ---- weight transpose + bf16 conversion (Wt[N][K]) ----
    dim3 tgrid(EE / 32, EE / 32);
    transpose_bf16_kernel<<<tgrid, 256, 0, stream>>>(wq, wqb);
    transpose_bf16_kernel<<<tgrid, 256, 0, stream>>>(wk, wkb);
    transpose_bf16_kernel<<<tgrid, 256, 0, stream>>>(wv, wvb);
    transpose_bf16_kernel<<<tgrid, 256, 0, stream>>>(wo, wob);

    // ---- RMSNorm 1 ----
    rmsnorm_bf16_kernel<<<dim3(TT, BB), 256, 0, stream>>>(x, ln1_w, xn);

    dim3 ggrid(EE / GTN, ROWS / GTM);

    // ---- Q/K/V projections + RoPE / transpose ----
    gemm_bf16_wmma_kernel<<<ggrid, 256, 0, stream>>>(xn, wqb, bq, tmp, ROWS, EE, EE);
    rope_transpose_kernel<<<dim3(TT, BB), 512, 0, stream>>>(tmp, qT, 1);
    gemm_bf16_wmma_kernel<<<ggrid, 256, 0, stream>>>(xn, wkb, bk, tmp, ROWS, EE, EE);
    rope_transpose_kernel<<<dim3(TT, BB), 512, 0, stream>>>(tmp, kT, 1);
    gemm_bf16_wmma_kernel<<<ggrid, 256, 0, stream>>>(xn, wvb, bv, tmp, ROWS, EE, EE);
    rope_transpose_kernel<<<dim3(TT, BB), 512, 0, stream>>>(tmp, vT, 0);

    // ---- Nystrom attention ----
    landmark_kernel<<<dim3(MM, HH, BB), 64, 0, stream>>>(qT, ql);
    landmark_kernel<<<dim3(MM, HH, BB), 64, 0, stream>>>(kT, kl);
    f3v_kernel<<<dim3(MM, HH, BB), 256, 0, stream>>>(ql, kT, vT, G);
    pinv_kernel<<<dim3(HH, BB), 256, 0, stream>>>(ql, kl, G, P, pscr);
    f1p_kernel<<<dim3(TT / 256, HH, BB), 256, 0, stream>>>(qT, kl, P, attnC);

    // ---- output projection ----
    gemm_bf16_wmma_kernel<<<ggrid, 256, 0, stream>>>(attnC, wob, bo, tmp, ROWS, EE, EE);

    // ---- gate + residual mix ----
    gate_kernel<<<dim3(TT, BB), 512, 0, stream>>>(x, tmp, alpha, gthetas, gbias, x2);

    // ---- FFN butterfly + residual ----
    ffn_kernel<<<dim3(TT, BB), 512, 0, stream>>>(x2, ln2_w, fthetas, out);
}
